// VQVAE_18889266168352
// MI455X (gfx1250) — matrix operand order repaired and checked by reading, not compiled, not run on hardware
//
#include <hip/hip_runtime.h>
#include <hip/hip_bf16.h>

typedef __attribute__((ext_vector_type(16))) _Float16 v16h;
typedef __attribute__((ext_vector_type(8)))  _Float16 v8h;
typedef __attribute__((ext_vector_type(8)))  float    v8f;
typedef __attribute__((ext_vector_type(4)))  uint32_t v4u;
typedef __attribute__((ext_vector_type(8)))  int      v8i;
typedef __attribute__((ext_vector_type(4)))  int      v4i;

#define LRELU 0.3f

#if defined(__HIP_DEVICE_COMPILE__) && __has_builtin(__builtin_amdgcn_tensor_load_to_lds)
#define HAVE_TDM 1
#else
#define HAVE_TDM 0
#endif

#if defined(__HIP_DEVICE_COMPILE__) && __has_builtin(__builtin_amdgcn_s_wait_tensorcnt)
#define WAIT_TENSOR(n) __builtin_amdgcn_s_wait_tensorcnt(n)
#elif defined(__HIP_DEVICE_COMPILE__)
#define WAIT_TENSOR(n) asm volatile("s_wait_tensorcnt %0" ::"i"(n) : "memory")
#else
#define WAIT_TENSOR(n)
#endif

// ---- CDNA5 wave32 WMMA fragment loaders (16x16x32 f16, contiguous k = 16*(lane/16) + s, same order in A and B) ----
// A (16x32, MxK): lanes 0-15 row M=lane, K 0..15; lanes 16-31 row M=lane-16, K 16..31.
__device__ __forceinline__ v16h frag_a(const _Float16* p, int stride) {
  int lane = threadIdx.x & 31;
  int row  = lane & 15;
  int kb   = (lane & 16) ? 16 : 0;
  const _Float16* rp = p + row * stride + kb;
  v16h a;
#pragma unroll
  for (int j = 0; j < 4; ++j) {
    a[2*j]     = rp[2*j];
    a[2*j+1]   = rp[2*j+1];
    a[8+2*j]   = rp[8+2*j];
    a[8+2*j+1] = rp[8+2*j+1];
  }
  return a;
}

// B (32x16, KxN) staged in LDS column-major [n][k] (f16 path, conv kernel).
__device__ __forceinline__ v16h frag_b(const _Float16* p, int stride) {
  int lane = threadIdx.x & 31;
  int n    = lane & 15;
  int kb   = (lane & 16) ? 16 : 0;
  const _Float16* rp = p + n * stride + kb;
  v16h b;
#pragma unroll
  for (int j = 0; j < 8; ++j) { b[2*j] = rp[2*j]; b[2*j+1] = rp[2*j+1]; }
  return b;
}

// B fragment from f32 LDS tile [k][129] (TDM-staged, memory order), n = global col in tile.
__device__ __forceinline__ v16h frag_b_f32(const float* p, int n) {
  int lane = threadIdx.x & 31;
  int kb   = (lane & 16) ? 16 : 0;
  v16h b;
#pragma unroll
  for (int j = 0; j < 16; ++j) b[j] = (_Float16)p[(kb + j) * 129 + n];
  return b;
}

#if HAVE_TDM
// Issue a TDM 2-D tile load: tile_rows x tile_cols f32 from gsrc (row stride =
// row_stride_elems) into LDS at lds_off, with 1-dword pad every 128 dwords so
// the LDS image is [tile_rows][129] f32.  Uniform args; wave-level op.
// 6-arg builtin form (clang-23 / therock-10.0): groups 2/3 + extra block are
// zero/NULL for a <=2-D tile.
__device__ __forceinline__ void tdm_load_2d_f32(uint32_t lds_off, const void* gsrc,
                                                int tile_cols, int tile_rows,
                                                int row_stride_elems) {
  uint64_t ga = (uint64_t)(uintptr_t)gsrc;
  v4u g0;
  g0[0] = 1u;                                            // count=1 valid user D#
  g0[1] = lds_off;                                       // lds_addr (bytes)
  g0[2] = (uint32_t)ga;                                  // global_addr[31:0]
  g0[3] = (uint32_t)((ga >> 32) & 0x01FFFFFFu) | 0x80000000u;  // addr[56:32] | type=2
  v8i g1;
  g1[0] = (int)((2u << 16) | (1u << 20) | (6u << 22));   // data_size=4B, pad_en, interval=128dw, amount=1dw
  g1[1] = (int)((uint32_t)tile_cols << 16);              // tensor_dim0[15:0] @ bits63:48
  g1[2] = (int)((((uint32_t)tile_cols) >> 16) | ((uint32_t)tile_rows << 16)); // dim0 hi | dim1 lo
  g1[3] = (int)((((uint32_t)tile_rows) >> 16) | ((uint32_t)tile_cols << 16)); // dim1 hi | tile_dim0
  g1[4] = tile_rows & 0xFFFF;                            // tile_dim1 (tile_dim2 = 0)
  g1[5] = row_stride_elems;                              // tensor_dim0_stride low32
  g1[6] = 0;
  g1[7] = 0;
  v4i gz4 = {0, 0, 0, 0};
  v8i gz8 = {0, 0, 0, 0, 0, 0, 0, 0};
  __builtin_amdgcn_tensor_load_to_lds(g0, g1, gz4, gz4, gz8, 0);
}
#endif

// ---------------- init: z = bias (for split-K atomics), stats = 0, loss = 0 ----
__global__ __launch_bounds__(256) void k_init(float* __restrict__ z,
                                              const float* __restrict__ enc_bd,
                                              float* __restrict__ stats,
                                              float* __restrict__ loss) {
  int tid = threadIdx.x;
  for (int i = tid; i < 4096; i += 256) z[i] = enc_bd[i & 127];
  for (int i = tid; i < 512; i += 256) stats[i] = 0.f;
  if (tid == 0) *loss = 0.f;
}

// ---------------- conv1: (32,256,256,1) -s2-> (32,128,128,64), relu, bn1 stats ----
__global__ __launch_bounds__(256) void k_conv1(const float* __restrict__ x,
                                               const float* __restrict__ w,
                                               const float* __restrict__ b,
                                               _Float16* __restrict__ out,
                                               float* __restrict__ st_sum,
                                               float* __restrict__ st_sq) {
  __shared__ float lw[576];
  __shared__ float lb[64];
  __shared__ float lstats[128];
  int tid = threadIdx.x;
  for (int e = tid; e < 576; e += 256) lw[e] = w[e];
  if (tid < 64) lb[tid] = b[tid];
  if (tid < 128) lstats[tid] = 0.f;
  __syncthreads();

  int m = blockIdx.x * 256 + tid;          // 32*128*128 pixels
  int n = m >> 14, rem = m & 16383;
  int oh = rem >> 7, ow = rem & 127;
  float tv[9];
#pragma unroll
  for (int kh = 0; kh < 3; ++kh)
#pragma unroll
    for (int kw = 0; kw < 3; ++kw) {
      int ih = 2 * oh + kh, iw = 2 * ow + kw;   // SAME s2: pad_lo=0
      tv[kh*3+kw] = (ih < 256 && iw < 256) ? x[(size_t)(n*256 + ih)*256 + iw] : 0.f;
    }
  for (int c = 0; c < 64; ++c) {
    float a = lb[c];
#pragma unroll
    for (int t = 0; t < 9; ++t) a += tv[t] * lw[t*64 + c];
    a = fmaxf(a, 0.f);
    out[(size_t)m * 64 + c] = (_Float16)a;
    atomicAdd(&lstats[c], a);
    atomicAdd(&lstats[64 + c], a * a);
  }
  __syncthreads();
  if (tid < 64) { atomicAdd(&st_sum[tid], lstats[tid]); atomicAdd(&st_sq[tid], lstats[64+tid]); }
}

// ---------------- bn finalize: sums -> per-channel scale/shift ----
__global__ void k_bnfinal(const float* __restrict__ sum, const float* __restrict__ sq,
                          const float* __restrict__ g, const float* __restrict__ b,
                          float cnt, float* __restrict__ scale, float* __restrict__ shift,
                          int nch) {
  int i = threadIdx.x;
  if (i < nch) {
    float m = sum[i] / cnt;
    float v = sq[i] / cnt - m * m;
    float s = g[i] * rsqrtf(v + 1e-3f);
    scale[i] = s;
    shift[i] = b[i] - m * s;
  }
}

// mode bits for k_conv_wmma
#define CM_T      1
#define CM_RELU   2
#define CM_SIG3   4
#define CM_STATS  8
#define CM_XF    16

// ---------------- generic implicit-GEMM conv / convT with WMMA ----------------
__global__ __launch_bounds__(256) void k_conv_wmma(
    const _Float16* __restrict__ in, const float* __restrict__ w,
    const float* __restrict__ bias,
    _Float16* __restrict__ out_h, float* __restrict__ out_f,
    const float* __restrict__ xf_scale, const float* __restrict__ xf_shift,
    float* __restrict__ st_sum, float* __restrict__ st_sq,
    int inH, int inW, int ci, int co, int outH, int outW, int mode) {
  __shared__ __attribute__((aligned(32))) _Float16 ldsA[8][16*40];
  __shared__ __attribute__((aligned(32))) _Float16 ldsB[64*40];
  __shared__ float lstats[128];

  const int tid = threadIdx.x, wave = tid >> 5, lane = tid & 31;
  const bool T = mode & CM_T, RELU = mode & CM_RELU, SIG = mode & CM_SIG3;
  const bool STATS = mode & CM_STATS, XF = mode & CM_XF;
  const int ntiles = (co + 15) >> 4;
  const int coPad = ntiles << 4;
  const int cchunks = ci >> 5;

  if (STATS && tid < 2 * co) lstats[tid] = 0.f;

  const int arow = lane & 15, ahalf = lane >> 4;
  const int m_a = blockIdx.x * 128 + wave * 16 + arow;
  const int HW = outH * outW;
  const int an = m_a / HW;
  const int arem = m_a - an * HW;
  const int aoh = arem / outW, aow = arem - (arem / outW) * outW;

  v8f acc[4] = {};

  for (int kh = 0; kh < 3; ++kh)
    for (int kw = 0; kw < 3; ++kw) {
      int ih, iw; bool ok;
      if (T) {            // out = 2*in + k - 1 (SAME convT, s2)
        int t = aoh + 1 - kh, u = aow + 1 - kw;
        ok = (t >= 0) && (u >= 0) && !((t | u) & 1);
        ih = t >> 1; iw = u >> 1;
        ok = ok && ih < inH && iw < inW;
      } else {            // in = 2*out + k (SAME fwd, s2, pad_lo=0)
        ih = 2 * aoh + kh; iw = 2 * aow + kw;
        ok = ih < inH && iw < inW;
      }
      for (int cc = 0; cc < cchunks; ++cc) {
        const int cbase = cc * 32 + ahalf * 16;
        v16h av = {};
        if (ok) av = *(const v16h*)(in + ((size_t)(an * inH + ih) * inW + iw) * ci + cbase);
        if (XF) {
#pragma unroll
          for (int i = 0; i < 16; ++i) {
            float v = (float)av[i] * xf_scale[cbase + i] + xf_shift[cbase + i];
            av[i] = (_Float16)(v >= 0.f ? v : LRELU * v);
          }
        }
        {
          _Float16* dst = &ldsA[wave][arow * 40 + ahalf * 16];
          v8h lo, hi;
#pragma unroll
          for (int i = 0; i < 8; ++i) { lo[i] = av[i]; hi[i] = av[8 + i]; }
          *(v8h*)dst = lo;
          *(v8h*)(dst + 8) = hi;
        }
        for (int e = tid; e < coPad * 32; e += 256) {
          int n = e >> 5, kk = e & 31;
          float wv = (n < co) ? w[((size_t)((kh * 3 + kw) * ci) + cc * 32 + kk) * co + n] : 0.f;
          ldsB[n * 40 + kk] = (_Float16)wv;
        }
        __syncthreads();
        v16h fa = frag_a(&ldsA[wave][0], 40);
#pragma unroll
        for (int nt = 0; nt < 4; ++nt)
          if (nt < ntiles) {
            v16h fb = frag_b(&ldsB[nt * 16 * 40], 40);
            acc[nt] = __builtin_amdgcn_wmma_f32_16x16x32_f16(
                false, fa, false, fb, (short)0, acc[nt], false, false);
          }
        __syncthreads();
      }
    }

  const int nloc = lane & 15, moff = (lane >> 4) * 8;
#pragma unroll
  for (int nt = 0; nt < 4; ++nt)
    if (nt < ntiles) {
      int ch = nt * 16 + nloc;
      bool chok = ch < co;
      float bv = chok ? bias[ch] : 0.f;
      float lsum = 0.f, lsq = 0.f;
#pragma unroll
      for (int r = 0; r < 8; ++r) {
        float v = acc[nt][r] + bv;
        if (RELU) v = fmaxf(v, 0.f);
        int pix = blockIdx.x * 128 + wave * 16 + moff + r;
        if (SIG) {
          if (ch < 3) out_f[(size_t)pix * 3 + ch] = 1.f / (1.f + expf(-v));
        } else if (chok) {
          out_h[(size_t)pix * co + ch] = (_Float16)v;
          lsum += v; lsq += v * v;
        }
      }
      if (STATS && chok) { atomicAdd(&lstats[ch], lsum); atomicAdd(&lstats[co + ch], lsq); }
    }
  if (STATS) {
    __syncthreads();
    if (tid < co) { atomicAdd(&st_sum[tid], lstats[tid]); atomicAdd(&st_sq[tid], lstats[co + tid]); }
  }
}

// ---------------- small-M (32) dense GEMM with WMMA + TDM weight streaming ----
// grid.x = K splits (atomic accumulate), grid.y = N blocks of 128.
// mode: bit0 split-K atomic f32 out; bit1 per-channel affine+leaky on A load.
// B tile (32 x 128 f32) is DMA'd by the Tensor Data Mover into LDS as [32][129]
// (1-dword pad per row), double buffered so the DMA overlaps the WMMAs.
__global__ __launch_bounds__(256) void k_gemm_wmma(
    const _Float16* __restrict__ Ah, const float* __restrict__ Af,
    const float* __restrict__ W, const float* __restrict__ bias,
    const float* __restrict__ xf_scale, const float* __restrict__ xf_shift, int cimod,
    float* __restrict__ out_acc, _Float16* __restrict__ out_h,
    int K, int N, int kspan, int mode) {
  __shared__ __attribute__((aligned(32))) _Float16 ldsA[32 * 40];
  __shared__ __attribute__((aligned(16))) float ldsBf[2][32 * 129];
  const int tid = threadIdx.x, wave = tid >> 5, lane = tid & 31;
  const bool SPLIT = mode & 1, XF = mode & 2;
  const int k0blk = blockIdx.x * kspan;
  const int nbase = blockIdx.y * 128;
  const int nsteps = kspan >> 5;
  v8f acc[2] = {};

#if HAVE_TDM
  if (wave == 0)
    tdm_load_2d_f32((uint32_t)(uintptr_t)&ldsBf[0][0], W + (size_t)k0blk * N + nbase,
                    128, 32, N);
#endif

  for (int s = 0; s < nsteps; ++s) {
    const int k0 = k0blk + s * 32;
    const int buf = s & 1;
    // ---- stage A (32 x 32 f16) ----
    if (Ah) {
      for (int e = tid; e < 128; e += 256) {
        int m = e >> 2, k8 = (e & 3) * 8;
        v8h av = *(const v8h*)(Ah + (size_t)m * K + k0 + k8);
        if (XF) {
#pragma unroll
          for (int i = 0; i < 8; ++i) {
            int c = (k0 + k8 + i) & (cimod - 1);
            float v = (float)av[i] * xf_scale[c] + xf_shift[c];
            av[i] = (_Float16)(v >= 0.f ? v : LRELU * v);
          }
        }
        *(v8h*)(&ldsA[m * 40 + k8]) = av;
      }
    } else {
      for (int e = tid; e < 1024; e += 256) {
        int m = e >> 5, kk = e & 31;
        ldsA[m * 40 + kk] = (_Float16)Af[(size_t)m * K + k0 + kk];
      }
    }
    // ---- B tile: TDM double-buffer, or plain-load fallback ----
#if HAVE_TDM
    if (wave == 0) {
      if (s + 1 < nsteps) {
        tdm_load_2d_f32((uint32_t)(uintptr_t)&ldsBf[buf ^ 1][0],
                        W + (size_t)(k0 + 32) * N + nbase, 128, 32, N);
        WAIT_TENSOR(1);   // current buffer done, prefetch may remain in flight
      } else {
        WAIT_TENSOR(0);
      }
    }
#else
    for (int e = tid; e < 1024; e += 256) {          // 32 k-rows x 32 float4
      int kk = e >> 5, g = e & 31;
      float4 wv = *(const float4*)(W + (size_t)(k0 + kk) * N + nbase + 4 * g);
      ldsBf[buf][kk * 129 + 4 * g + 0] = wv.x;
      ldsBf[buf][kk * 129 + 4 * g + 1] = wv.y;
      ldsBf[buf][kk * 129 + 4 * g + 2] = wv.z;
      ldsBf[buf][kk * 129 + 4 * g + 3] = wv.w;
    }
#endif
    __syncthreads();
    v16h fb = frag_b_f32(&ldsBf[buf][0], wave * 16 + (lane & 15));
#pragma unroll
    for (int mt = 0; mt < 2; ++mt) {
      v16h fa = frag_a(&ldsA[mt * 16 * 40], 40);
      acc[mt] = __builtin_amdgcn_wmma_f32_16x16x32_f16(
          false, fa, false, fb, (short)0, acc[mt], false, false);
    }
    __syncthreads();
  }

  const int nloc = lane & 15, moff = (lane >> 4) * 8;
  const int ng = nbase + wave * 16 + nloc;
#pragma unroll
  for (int mt = 0; mt < 2; ++mt)
#pragma unroll
    for (int r = 0; r < 8; ++r) {
      int m = mt * 16 + moff + r;
      float v = acc[mt][r];
      if (SPLIT) atomicAdd(&out_acc[(size_t)m * N + ng], v);
      else out_h[(size_t)m * N + ng] = (_Float16)(v + bias[ng]);
    }
}

// ---------------- VQ: per row of zf (64x64), argmin over 512 codes, loss ----
__global__ __launch_bounds__(256) void k_vq(const float* __restrict__ z,
                                            const float* __restrict__ cb,
                                            float* __restrict__ zq,
                                            float* __restrict__ loss_out) {
  __shared__ float zrow[64];
  __shared__ float bd[256];
  __shared__ int   bi[256];
  __shared__ float lloss;
  int r = blockIdx.x, tid = threadIdx.x;
  if (tid < 64) zrow[tid] = z[r * 64 + tid];
  if (tid == 0) lloss = 0.f;
  __syncthreads();
  float best = 3.4e38f; int bidx = 0;
  for (int c = tid; c < 512; c += 256) {
    float d = 0.f;
    for (int k = 0; k < 64; ++k) { float t = zrow[k] - cb[(size_t)c * 64 + k]; d += t * t; }
    if (d < best) { best = d; bidx = c; }
  }
  bd[tid] = best; bi[tid] = bidx;
  __syncthreads();
  for (int s = 128; s > 0; s >>= 1) {
    if (tid < s && bd[tid + s] < bd[tid]) { bd[tid] = bd[tid + s]; bi[tid] = bi[tid + s]; }
    __syncthreads();
  }
  int idx = bi[0];
  if (tid < 64) {
    float q = cb[(size_t)idx * 64 + tid];
    zq[(size_t)r * 64 + tid] = q;
    float d = q - zrow[tid];
    atomicAdd(&lloss, d * d);
  }
  __syncthreads();
  if (tid == 0) atomicAdd(loss_out, lloss * 1.25f / 4096.0f);  // (1+0.25)*mean
}

extern "C" void kernel_launch(void* const* d_in, const int* in_sizes, int n_in,
                              void* d_out, int out_size, void* d_ws, size_t ws_size,
                              hipStream_t stream) {
  const float* x        = (const float*)d_in[0];
  const float* enc_w1   = (const float*)d_in[1];
  const float* enc_b1   = (const float*)d_in[2];
  const float* bn1_g    = (const float*)d_in[3];
  const float* bn1_b    = (const float*)d_in[4];
  const float* enc_w2   = (const float*)d_in[5];
  const float* enc_b2   = (const float*)d_in[6];
  const float* bn2_g    = (const float*)d_in[7];
  const float* bn2_b    = (const float*)d_in[8];
  const float* enc_wd   = (const float*)d_in[9];
  const float* enc_bd   = (const float*)d_in[10];
  const float* codebook = (const float*)d_in[11];
  const float* dec_wd   = (const float*)d_in[12];
  const float* dec_bd   = (const float*)d_in[13];
  const float* dec_w1   = (const float*)d_in[14];
  const float* dec_b1   = (const float*)d_in[15];
  const float* dec_w2   = (const float*)d_in[16];
  const float* dec_b2   = (const float*)d_in[17];
  const float* dbn1_g   = (const float*)d_in[18];
  const float* dbn1_b   = (const float*)d_in[19];
  const float* dec_w3   = (const float*)d_in[20];
  const float* dec_b3   = (const float*)d_in[21];
  const float* dec_w4   = (const float*)d_in[22];
  const float* dec_b4   = (const float*)d_in[23];
  const float* dbn2_g   = (const float*)d_in[24];
  const float* dbn2_b   = (const float*)d_in[25];
  const float* dec_w5   = (const float*)d_in[26];
  const float* dec_b5   = (const float*)d_in[27];

  char* ws = (char*)d_ws;
  size_t off = 0;
  auto take = [&](size_t bytes) { char* p = ws + off; off = (off + bytes + 255) & ~(size_t)255; return p; };

  _Float16* h1 = (_Float16*)take((size_t)32*128*128*64*2);
  _Float16* h2 = (_Float16*)take((size_t)32*64*64*64*2);
  _Float16* h3 = (_Float16*)take((size_t)32*8*8*64*2);
  _Float16* h4 = (_Float16*)take((size_t)32*16*16*32*2);
  _Float16* h5 = (_Float16*)take((size_t)32*32*32*32*2);
  _Float16* h6 = (_Float16*)take((size_t)32*64*64*32*2);
  _Float16* h7 = (_Float16*)take((size_t)32*128*128*32*2);
  float* z     = (float*)take(4096 * 4);
  float* zq    = (float*)take(4096 * 4);
  float* stats = (float*)take(512 * 4);
  float* aff   = (float*)take(512 * 4);

  float* loss = (float*)d_out;
  float* xhat = (float*)d_out + 1;

  k_init<<<1, 256, 0, stream>>>(z, enc_bd, stats, loss);

  // encoder
  k_conv1<<<2048, 256, 0, stream>>>(x, enc_w1, enc_b1, h1, stats + 0, stats + 64);
  k_bnfinal<<<1, 64, 0, stream>>>(stats + 0, stats + 64, bn1_g, bn1_b, 524288.f, aff + 0, aff + 64, 64);
  k_conv_wmma<<<1024, 256, 0, stream>>>(h1, enc_w2, enc_b2, h2, nullptr,
      aff + 0, aff + 64, stats + 128, stats + 192,
      128, 128, 64, 64, 64, 64, CM_RELU | CM_STATS | CM_XF);
  k_bnfinal<<<1, 64, 0, stream>>>(stats + 128, stats + 192, bn2_g, bn2_b, 131072.f, aff + 128, aff + 192, 64);
  // z = leaky(bn2(h2)) @ enc_wd  (split-K, atomic accumulate onto bias-initialized z)
  k_gemm_wmma<<<dim3(256, 1), 256, 0, stream>>>(h2, nullptr, enc_wd, nullptr,
      aff + 128, aff + 192, 64, z, nullptr, 262144, 128, 1024, 1 | 2);

  // vector quantization + embedding loss
  k_vq<<<64, 256, 0, stream>>>(z, codebook, zq, loss);

  // decoder
  k_gemm_wmma<<<dim3(1, 32), 256, 0, stream>>>(nullptr, zq, dec_wd, dec_bd,
      nullptr, nullptr, 64, nullptr, h3, 128, 4096, 128, 0);
  k_conv_wmma<<<64, 256, 0, stream>>>(h3, dec_w1, dec_b1, h4, nullptr,
      nullptr, nullptr, nullptr, nullptr, 8, 8, 64, 32, 16, 16, CM_T | CM_RELU);
  k_conv_wmma<<<256, 256, 0, stream>>>(h4, dec_w2, dec_b2, h5, nullptr,
      nullptr, nullptr, stats + 256, stats + 288, 16, 16, 32, 32, 32, 32, CM_T | CM_RELU | CM_STATS);
  k_bnfinal<<<1, 64, 0, stream>>>(stats + 256, stats + 288, dbn1_g, dbn1_b, 32768.f, aff + 256, aff + 288, 32);
  k_conv_wmma<<<1024, 256, 0, stream>>>(h5, dec_w3, dec_b3, h6, nullptr,
      aff + 256, aff + 288, nullptr, nullptr, 32, 32, 32, 32, 64, 64, CM_T | CM_RELU | CM_XF);
  k_conv_wmma<<<4096, 256, 0, stream>>>(h6, dec_w4, dec_b4, h7, nullptr,
      nullptr, nullptr, stats + 320, stats + 352, 64, 64, 32, 32, 128, 128, CM_T | CM_RELU | CM_STATS);
  k_bnfinal<<<1, 64, 0, stream>>>(stats + 320, stats + 352, dbn2_g, dbn2_b, 524288.f, aff + 320, aff + 352, 32);
  k_conv_wmma<<<16384, 256, 0, stream>>>(h7, dec_w5, dec_b5, nullptr, xhat,
      aff + 320, aff + 352, nullptr, nullptr, 128, 128, 32, 3, 256, 256, CM_T | CM_SIG3 | CM_XF);
}
